// GraphSwAVModel_72559177499162
// MI455X (gfx1250) — matrix operand order repaired and checked by reading, not compiled, not run on hardware
//
#include <hip/hip_runtime.h>
#include <hip/hip_bf16.h>

// ---------------------------------------------------------------------------
// GraphSwAV (GINEConv + GlobalAttention + head) for MI455X / gfx1250.
// All dense GEMMs use v_wmma_f32_16x16x32_bf16 (wave32, 16x16 C tiles).
// Scatter segment-sums use global_atomic_add_f32 (no-return; agg[50000x128]
// = 25.6MB and x = 25.6MB stay resident in the 192MB L2, while the 205MB
// edge_attr stream is loaded non-temporally so it does not evict them).
// All device buffers are < 2GiB so addressing uses 32-bit byte offsets
// (SGPR base + VGPR offset form) to minimize VALU in the hot loops.
// ---------------------------------------------------------------------------

typedef __attribute__((ext_vector_type(16))) __bf16 v16bf;
typedef __attribute__((ext_vector_type(8)))  float  v8f;
typedef __attribute__((ext_vector_type(4)))  float  v4f;
typedef __bf16 bf16;

#define N_NODES 50000
#define N_EDGES 800000
#define IN_C    128
#define ED_C    64
#define HID_C   256
#define N_B     128
#define N_K     1000
#define N_KPAD  1024

// ---- WMMA helpers ----------------------------------------------------------

__device__ __forceinline__ v8f wmma_bf16(v16bf a, v16bf b, v8f c) {
  // D = A(16x32) * B(32x16) + C(16x16), fp32 accum
  return __builtin_amdgcn_wmma_f32_16x16x32_bf16(
      /*neg_a=*/false, a, /*neg_b=*/false, b,
      /*c_mod=*/(short)0, c, /*reuse_a=*/false, /*reuse_b=*/false);
}

__device__ __forceinline__ v16bf pack_a_frag(v4f f0, v4f f1, v4f f2, v4f f3) {
  v16bf r;
  r[0]=(bf16)f0[0]; r[1]=(bf16)f0[1]; r[2]=(bf16)f0[2]; r[3]=(bf16)f0[3];
  r[4]=(bf16)f1[0]; r[5]=(bf16)f1[1]; r[6]=(bf16)f1[2]; r[7]=(bf16)f1[3];
  r[8]=(bf16)f2[0]; r[9]=(bf16)f2[1]; r[10]=(bf16)f2[2]; r[11]=(bf16)f2[3];
  r[12]=(bf16)f3[0]; r[13]=(bf16)f3[1]; r[14]=(bf16)f3[2]; r[15]=(bf16)f3[3];
  return r;
}

// A fragment (contiguous k order, matching load_b_frag): lane l -> row m=l&15, half=l>>4.
// VGPR0-7 hold k = base..base+15,
// where base = s*32 + half*16.  `p` points at A + row*K + s*32 + half*16.
__device__ __forceinline__ v16bf load_a_frag_f32(const float* __restrict__ p) {
  return pack_a_frag(*(const v4f*)(p + 0),  *(const v4f*)(p + 4),
                     *(const v4f*)(p + 8), *(const v4f*)(p + 12));
}

// Non-temporal variant for single-use streaming data (edge_attr, 205MB).
__device__ __forceinline__ v16bf load_a_frag_f32_nt(const float* __restrict__ p) {
  v4f f0 = __builtin_nontemporal_load((const v4f*)(p + 0));
  v4f f1 = __builtin_nontemporal_load((const v4f*)(p + 4));
  v4f f2 = __builtin_nontemporal_load((const v4f*)(p + 8));
  v4f f3 = __builtin_nontemporal_load((const v4f*)(p + 12));
  return pack_a_frag(f0, f1, f2, f3);
}

// B fragment (16-bit B 32x16): lane l -> col n=l&15, kb = s*32 + (l>>4)*16.
// With W pre-transposed to [Nout][K] bf16, this is one contiguous 32B load.
__device__ __forceinline__ v16bf load_b_frag(const bf16* __restrict__ Wt,
                                             unsigned n, unsigned K, unsigned kb) {
  return *(const v16bf*)(Wt + n * K + kb);
}

// ---- small utility kernels -------------------------------------------------

__global__ void transpose_to_bf16(const float* __restrict__ src, // [K][Nsrc]
                                  bf16* __restrict__ dst,        // [Npad][K]
                                  int K, int Nsrc, int Npad) {
  int idx = blockIdx.x * blockDim.x + threadIdx.x;
  if (idx >= Npad * K) return;
  int n = idx / K, k = idx - n * K;
  dst[idx] = (n < Nsrc) ? (bf16)src[(unsigned)(k * Nsrc + n)] : (bf16)0.0f;
}

__global__ void copy_f32x4(const float* __restrict__ src, float* __restrict__ dst,
                           int n4) {
  int i = blockIdx.x * blockDim.x + threadIdx.x;
  if (i < n4) ((v4f*)dst)[i] = ((const v4f*)src)[i];
}

__global__ void fill_f32(float* p, float v, int n) {
  int i = blockIdx.x * blockDim.x + threadIdx.x;
  if (i < n) p[i] = v;
}
__global__ void fill_u32(unsigned* p, unsigned v, int n) {
  int i = blockIdx.x * blockDim.x + threadIdx.x;
  if (i < n) p[i] = v;
}

// monotonic float <-> uint mapping for atomic max
__device__ __forceinline__ unsigned fmap(float f) {
  unsigned u = __float_as_uint(f);
  return (u & 0x80000000u) ? ~u : (u | 0x80000000u);
}
__device__ __forceinline__ float funmap(unsigned m) {
  unsigned b = (m & 0x80000000u) ? (m & 0x7FFFFFFFu) : ~m;
  return __uint_as_float(b);
}

// ---- edge kernel: e = ea@We+be ; m = relu(x[src]+e) ; agg[dst] += m --------
// one wave handles 16 edges x all 128 outputs: 8 N-tiles x 2 K-steps = 16 WMMAs
__global__ void edge_kernel(const float* __restrict__ ea,   // [E][64]
                            const int* __restrict__ ei,     // [2][E]
                            const float* __restrict__ x,    // [N][128]
                            const bf16* __restrict__ We_t,  // [128][64]
                            const float* __restrict__ be,   // [128]
                            float* __restrict__ agg, int E) {
  int wave = (blockIdx.x * blockDim.x + threadIdx.x) >> 5;
  int lane = threadIdx.x & 31;
  if (wave >= (E >> 4)) return;               // wave-uniform: EXEC stays all-1s
  int e0 = wave << 4;
  int m = lane & 15, half = lane >> 4;

  const float* arow = ea + (size_t)(e0 + m) * ED_C + half * 16;
  v16bf a0 = load_a_frag_f32_nt(arow);        // k = 0..31  (streaming, TH=NT)
  v16bf a1 = load_a_frag_f32_nt(arow + 32);   // k = 32..63

  int rbase = e0 + half * 8;                  // output rows this lane owns
  unsigned sOff[8], dOff[8];                  // 32-bit byte offsets (<2^25)
#pragma unroll
  for (int i = 0; i < 8; ++i) {
    sOff[i] = ((unsigned)ei[rbase + i]) << 9;       // *IN_C*sizeof(float)
    dOff[i] = ((unsigned)ei[E + rbase + i]) << 9;
  }

  unsigned nl = lane & 15;
#pragma unroll
  for (int j = 0; j < 8; ++j) {
    unsigned n = j * 16 + nl;
    v8f acc = {0.f, 0.f, 0.f, 0.f, 0.f, 0.f, 0.f, 0.f};
    acc = wmma_bf16(a0, load_b_frag(We_t, n, ED_C, half * 16), acc);
    acc = wmma_bf16(a1, load_b_frag(We_t, n, ED_C, 32 + half * 16), acc);
    float bev = be[n];
    unsigned nb = n << 2;
#pragma unroll
    for (int i = 0; i < 8; ++i) {
      float xv = *(const float*)((const char*)x + (sOff[i] + nb));
      float v = acc[i] + bev + xv;
      v = v > 0.f ? v : 0.f;
      atomicAdd((float*)((char*)agg + (dOff[i] + nb)), v);
    }
  }
}

// ---- generic GEMM: C[M][Nreal] = act(A[M][K] @ W + bias) -------------------
// Wt is [Npad][K] bf16 (pre-transposed, zero-padded rows). One wave computes a
// 16x64 tile (A-frag reused across 4 N-subtiles).
template <int ACT>
__global__ void gemm_bias_kernel(const float* __restrict__ A,
                                 const bf16* __restrict__ Wt,
                                 const float* __restrict__ bias,
                                 float* __restrict__ C,
                                 int M, int K, int Npad, int Nreal) {
  int wave = (blockIdx.x * blockDim.x + threadIdx.x) >> 5;
  int lane = threadIdx.x & 31;
  int ngroups = Npad >> 6;
  if (wave >= (M >> 4) * ngroups) return;     // wave-uniform
  int mt = wave / ngroups, ng = wave - mt * ngroups;
  unsigned m0 = mt << 4, n0 = ng << 6;
  unsigned nl = lane & 15, half = lane >> 4;
  unsigned mrow = m0 + nl;

  const float* arow = A + mrow * (unsigned)K + half * 16;   // 32-bit offsets
  v8f z = {0.f, 0.f, 0.f, 0.f, 0.f, 0.f, 0.f, 0.f};
  v8f acc0 = z, acc1 = z, acc2 = z, acc3 = z;

  for (unsigned s = 0; s < (unsigned)K; s += 32) {
    v16bf af = load_a_frag_f32(arow + s);
    unsigned kb = s + half * 16;
    acc0 = wmma_bf16(af, load_b_frag(Wt, n0 + 0 + nl,  K, kb), acc0);
    acc1 = wmma_bf16(af, load_b_frag(Wt, n0 + 16 + nl, K, kb), acc1);
    acc2 = wmma_bf16(af, load_b_frag(Wt, n0 + 32 + nl, K, kb), acc2);
    acc3 = wmma_bf16(af, load_b_frag(Wt, n0 + 48 + nl, K, kb), acc3);
  }

  v8f accs[4] = {acc0, acc1, acc2, acc3};
#pragma unroll
  for (int j = 0; j < 4; ++j) {
    unsigned n = n0 + j * 16 + nl;
    if ((int)n < Nreal) {                      // divergence OK: no WMMA below
      float bv = bias[n];
#pragma unroll
      for (int i = 0; i < 8; ++i) {
        unsigned mm = m0 + half * 8 + i;
        float v = accs[j][i] + bv;
        if (ACT) v = v > 0.f ? v : 0.f;
        C[mm * (unsigned)Nreal + n] = v;
      }
    }
  }
}

// ---- gate = g1 @ Wg2 + bg2 ; per-graph running max -------------------------
__global__ void gate_kernel(const float* __restrict__ g1,   // [N][128]
                            const float* __restrict__ Wg2,  // [128]
                            const float* __restrict__ bg2,  // [1]
                            const int* __restrict__ batch,
                            float* __restrict__ gate,
                            unsigned* __restrict__ gmax, int N) {
  int n = blockIdx.x * blockDim.x + threadIdx.x;
  if (n >= N) return;
  float acc = bg2[0];
  const v4f* gp = (const v4f*)(g1 + (unsigned)n * 128u);
  const v4f* wp = (const v4f*)Wg2;
#pragma unroll 8
  for (int k = 0; k < 32; ++k) {
    v4f a = gp[k], w = wp[k];
    acc += a[0] * w[0] + a[1] * w[1] + a[2] * w[2] + a[3] * w[3];
  }
  gate[n] = acc;
  atomicMax(gmax + batch[n], fmap(acc));
}

// ---- a = exp(gate - gmax[b]) ; denom[b] += a -------------------------------
__global__ void exp_kernel(const float* __restrict__ gate,
                           const unsigned* __restrict__ gmax,
                           const int* __restrict__ batch,
                           float* __restrict__ a, float* __restrict__ denom,
                           int N) {
  int n = blockIdx.x * blockDim.x + threadIdx.x;
  if (n >= N) return;
  int b = batch[n];
  float v = __expf(gate[n] - funmap(gmax[b]));
  a[n] = v;
  atomicAdd(denom + b, v);
}

// ---- pooled[b] += (a[n]/denom[b]) * h[n] -----------------------------------
__global__ void pool_kernel(const float* __restrict__ a,
                            const float* __restrict__ denom,
                            const int* __restrict__ batch,
                            const float* __restrict__ h,    // [N][256]
                            float* __restrict__ pooled,     // [B][256]
                            int N) {
  int idx = blockIdx.x * blockDim.x + threadIdx.x;
  if (idx >= N * 64) return;
  unsigned n = (unsigned)idx >> 6, c = ((unsigned)idx & 63u) << 2;
  int b = batch[n];
  float attn = a[n] / denom[b];
  v4f hv = *(const v4f*)(h + n * (unsigned)HID_C + c);
  float* pb = pooled + (unsigned)b * HID_C + c;
  atomicAdd(pb + 0, attn * hv[0]);
  atomicAdd(pb + 1, attn * hv[1]);
  atomicAdd(pb + 2, attn * hv[2]);
  atomicAdd(pb + 3, attn * hv[3]);
}

// ---------------------------------------------------------------------------

static inline int nblk(long long threads, int bs) {
  return (int)((threads + bs - 1) / bs);
}

extern "C" void kernel_launch(void* const* d_in, const int* in_sizes, int n_in,
                              void* d_out, int out_size, void* d_ws, size_t ws_size,
                              hipStream_t stream) {
  const float* x     = (const float*)d_in[0];
  const int*   ei    = (const int*)d_in[1];
  const float* ea    = (const float*)d_in[2];
  const int*   batch = (const int*)d_in[3];
  const float* We  = (const float*)d_in[4];
  const float* be  = (const float*)d_in[5];
  const float* W1  = (const float*)d_in[6];
  const float* b1  = (const float*)d_in[7];
  const float* W2  = (const float*)d_in[8];
  const float* b2  = (const float*)d_in[9];
  const float* Wg1 = (const float*)d_in[10];
  const float* bg1 = (const float*)d_in[11];
  const float* Wg2 = (const float*)d_in[12];
  const float* bg2 = (const float*)d_in[13];
  const float* Wh  = (const float*)d_in[14];
  const float* bh  = (const float*)d_in[15];

  char* ws = (char*)d_ws;
  size_t off = 0;
  auto alloc = [&](size_t bytes) {
    size_t o = off;
    off = (off + bytes + 255) & ~(size_t)255;
    return o;
  };
  float*    agg   = (float*)(ws + alloc((size_t)N_NODES * IN_C * 4));   // also g1
  float*    h1    = (float*)(ws + alloc((size_t)N_NODES * HID_C * 4));
  float*    h     = (float*)(ws + alloc((size_t)N_NODES * HID_C * 4));
  float*    gate  = (float*)(ws + alloc((size_t)N_NODES * 4));
  float*    avec  = (float*)(ws + alloc((size_t)N_NODES * 4));
  unsigned* gmax  = (unsigned*)(ws + alloc(N_B * 4));
  float*    denom = (float*)(ws + alloc(N_B * 4));
  bf16* We_t  = (bf16*)(ws + alloc((size_t)IN_C  * ED_C  * 2));  // [128][64]
  bf16* W1_t  = (bf16*)(ws + alloc((size_t)HID_C * IN_C  * 2));  // [256][128]
  bf16* W2_t  = (bf16*)(ws + alloc((size_t)HID_C * HID_C * 2));  // [256][256]
  bf16* Wg1_t = (bf16*)(ws + alloc((size_t)IN_C  * HID_C * 2));  // [128][256]
  bf16* Wh_t  = (bf16*)(ws + alloc((size_t)N_KPAD * HID_C * 2)); // [1024][256]
  (void)ws_size; (void)in_sizes; (void)n_in; (void)out_size;

  float* logits = (float*)d_out;                 // [128][1000]
  float* pooled = logits + (size_t)N_B * N_K;    // [128][256], also an output

  const int BS = 256;

  // 1) weight transposes to bf16 [Nout][K]
  transpose_to_bf16<<<nblk((size_t)IN_C * ED_C, BS), BS, 0, stream>>>(We, We_t, ED_C, IN_C, IN_C);
  transpose_to_bf16<<<nblk((size_t)HID_C * IN_C, BS), BS, 0, stream>>>(W1, W1_t, IN_C, HID_C, HID_C);
  transpose_to_bf16<<<nblk((size_t)HID_C * HID_C, BS), BS, 0, stream>>>(W2, W2_t, HID_C, HID_C, HID_C);
  transpose_to_bf16<<<nblk((size_t)IN_C * HID_C, BS), BS, 0, stream>>>(Wg1, Wg1_t, HID_C, IN_C, IN_C);
  transpose_to_bf16<<<nblk((size_t)N_KPAD * HID_C, BS), BS, 0, stream>>>(Wh, Wh_t, HID_C, N_K, N_KPAD);

  // 2) init: agg = x (so agg ends as x + segment_sum(m)); softmax scratch; pooled=0
  copy_f32x4<<<nblk((size_t)N_NODES * IN_C / 4, BS), BS, 0, stream>>>(x, agg, N_NODES * IN_C / 4);
  fill_u32<<<1, 256, 0, stream>>>(gmax, 0x00800000u /* fmap(-FLT_MAX) */, N_B);
  fill_f32<<<1, 256, 0, stream>>>(denom, 0.f, N_B);
  fill_f32<<<nblk(N_B * HID_C, BS), BS, 0, stream>>>(pooled, 0.f, N_B * HID_C);

  // 3) GINE message + scatter-add: 50000 tiles (waves) of 16 edges
  edge_kernel<<<(N_EDGES / 16) / 8, BS, 0, stream>>>(ea, ei, x, We_t, be, agg, N_EDGES);

  // 4) h1 = relu(hpre @ W1 + b1)   [50000x128]@[128x256]
  gemm_bias_kernel<1><<<nblk((long long)(N_NODES / 16) * (HID_C / 64) * 32, BS), BS, 0, stream>>>(
      agg, W1_t, b1, h1, N_NODES, IN_C, HID_C, HID_C);
  // 5) h = h1 @ W2 + b2            [50000x256]@[256x256]
  gemm_bias_kernel<0><<<nblk((long long)(N_NODES / 16) * (HID_C / 64) * 32, BS), BS, 0, stream>>>(
      h1, W2_t, b2, h, N_NODES, HID_C, HID_C, HID_C);
  // 6) g1 = relu(h @ Wg1 + bg1)    [50000x256]@[256x128]  (reuses agg buffer)
  gemm_bias_kernel<1><<<nblk((long long)(N_NODES / 16) * (IN_C / 64) * 32, BS), BS, 0, stream>>>(
      h, Wg1_t, bg1, agg, N_NODES, HID_C, IN_C, IN_C);

  // 7) gate + per-graph max, 8) exp + denom, 9) pooled
  gate_kernel<<<nblk(N_NODES, BS), BS, 0, stream>>>(agg, Wg2, bg2, batch, gate, gmax, N_NODES);
  exp_kernel<<<nblk(N_NODES, BS), BS, 0, stream>>>(gate, gmax, batch, avec, denom, N_NODES);
  pool_kernel<<<nblk((long long)N_NODES * 64, BS), BS, 0, stream>>>(avec, denom, batch, h, pooled, N_NODES);

  // 10) logits = pooled @ Wh + bh  [128x256]@[256x1000] (N padded to 1024)
  gemm_bias_kernel<0><<<nblk((long long)(N_B / 16) * (N_KPAD / 64) * 32, BS), BS, 0, stream>>>(
      pooled, Wh_t, bh, logits, N_B, HID_C, N_KPAD, N_K);
}
